// GatedAttnAggregator_35699768164475
// MI455X (gfx1250) — matrix operand order repaired and checked by reading, not compiled, not run on hardware
//
#include <hip/hip_runtime.h>
#include <hip/hip_bf16.h>

// ---------------------------------------------------------------------------
// GatedAttnAggregator for MI455X (gfx1250), fp32 in/out, bf16 WMMA GEMMs.
//
// Fusion:  energies[n,k] = X[n,k,:]·(Q[n]@Wk) + bk·Q[n]
//          context[n]    = (Σ_k att·X[n,k,:]) @ Wv^T + bv
// so no [N*K,256] GEMMs and no materialized keys/values.
// ---------------------------------------------------------------------------

#define N_NODES 16384
#define KNBR    32
#define DIN     256
#define DOUT    256
#define DCAT    512
#define BN_EPS  1e-5f

typedef __attribute__((ext_vector_type(16))) __bf16         bf16x16;
typedef __attribute__((ext_vector_type(8)))  float          floatx8;
typedef __attribute__((ext_vector_type(8)))  unsigned short ushort8;

union Frag16 { bf16x16 v; ushort8 h[2]; };

__device__ __forceinline__ unsigned short f2bf(float f) {
  unsigned int u = __float_as_uint(f);
  unsigned int r = u + 0x7FFFu + ((u >> 16) & 1u);   // round-to-nearest-even
  return (unsigned short)(r >> 16);
}

// ---------------------------------------------------------------------------
// Tiled bf16 WMMA GEMM:  C[M, Nc](+colOff, ldc) = act( A[M,K] * B^T + bias )
//   TRANSB=true : B stored [Nc,K] row-major (i.e. C = A @ W^T, torch Linear)
//   TRANSB=false: B stored [K,Nc] row-major (i.e. C = A @ B)
// Block: 256 threads = 8 waves in a 4(M) x 2(N) grid -> 64x32 macro tile.
// ---------------------------------------------------------------------------
template <bool TRANSB, bool RELU>
__global__ void gemm_bf16_wmma(const float* __restrict__ A, int lda,
                               const float* __restrict__ B,
                               const float* __restrict__ bias,
                               float* __restrict__ C, int ldc, int colOff,
                               int Nc, int K) {
  constexpr int BM = 64, BN = 32, BK = 32;
  constexpr int APAD = 40, BPAD = 40;   // element stride: 80B rows -> no bank conflicts
  __shared__ __align__(16) unsigned short As[BM * APAD];
  __shared__ __align__(16) unsigned short Bs[BN * BPAD];

  const int t    = threadIdx.x;       // 0..255
  const int lane = t & 31;
  const int wave = t >> 5;            // 0..7
  const int wm   = wave >> 1;         // 0..3  (16-row subtile)
  const int wn   = wave & 1;          // 0..1  (16-col subtile)
  const int half = lane >> 4;         // 0/1
  const int l15  = lane & 15;

  const int rowBase = blockIdx.x * BM;
  const int colBase = blockIdx.y * BN;

  floatx8 acc = {0.f, 0.f, 0.f, 0.f, 0.f, 0.f, 0.f, 0.f};

  for (int kc = 0; kc < K; kc += BK) {
    // ---- stage A tile (64x32 fp32 -> bf16), 8 elems/thread -----------------
    {
      const int i = t * 8;
      const int r = i >> 5;
      const int k = i & 31;           // 0,8,16,24
      const float* src = A + (size_t)(rowBase + r) * lda + kc + k;
      float4 f0 = *(const float4*)(src);
      float4 f1 = *(const float4*)(src + 4);
      unsigned short* dst = As + r * APAD + k;
      dst[0] = f2bf(f0.x); dst[1] = f2bf(f0.y); dst[2] = f2bf(f0.z); dst[3] = f2bf(f0.w);
      dst[4] = f2bf(f1.x); dst[5] = f2bf(f1.y); dst[6] = f2bf(f1.z); dst[7] = f2bf(f1.w);
      if (kc + BK < K) __builtin_prefetch(src + BK, 0, 0);   // global_prefetch_b8
    }
    // ---- stage B tile (32x32 -> Bs[n][k]) ---------------------------------
    if (TRANSB) {
      const int i = t * 4;
      const int n = i >> 5;
      const int k = i & 31;           // multiple of 4
      float4 f = *(const float4*)(B + (size_t)(colBase + n) * K + kc + k);
      unsigned short* dst = Bs + n * BPAD + k;
      dst[0] = f2bf(f.x); dst[1] = f2bf(f.y); dst[2] = f2bf(f.z); dst[3] = f2bf(f.w);
    } else {
      const int i = t * 4;
      const int k = i >> 5;
      const int n = i & 31;           // multiple of 4
      float4 f = *(const float4*)(B + (size_t)(kc + k) * Nc + colBase + n);
      Bs[(n + 0) * BPAD + k] = f2bf(f.x);
      Bs[(n + 1) * BPAD + k] = f2bf(f.y);
      Bs[(n + 2) * BPAD + k] = f2bf(f.z);
      Bs[(n + 3) * BPAD + k] = f2bf(f.w);
    }
    __syncthreads();

    // ---- fragments per documented gfx1250 16-bit layouts -------------------
    // A 16x32: lane l<16 holds row l, K={0..7,16..23}; lane l+16: K={8..15,24..31}
    Frag16 a, b;
    const unsigned short* arow = As + (wm * 16 + l15) * APAD;
    a.h[0] = *(const ushort8*)(arow + half * 8);
    a.h[1] = *(const ushort8*)(arow + 16 + half * 8);
    // B 32x16: lane l holds column l&15, K same as A: {0..7,16..23} / {8..15,24..31}
    const unsigned short* brow = Bs + (wn * 16 + l15) * BPAD;
    b.h[0] = *(const ushort8*)(brow + half * 8);
    b.h[1] = *(const ushort8*)(brow + 16 + half * 8);

    acc = __builtin_amdgcn_wmma_f32_16x16x32_bf16(false, a.v, false, b.v,
                                                  (short)0, acc, false, false);
    __syncthreads();
  }

  // ---- writeback: VGPR r -> M = r + 8*half, N = l15 ------------------------
  const int col = colBase + wn * 16 + l15;
  const float bsv = bias ? bias[col] : 0.f;
#pragma unroll
  for (int r = 0; r < 8; ++r) {
    const int row = rowBase + wm * 16 + r + 8 * half;
    float v = acc[r] + bsv;
    if (RELU) v = fmaxf(v, 0.f);
    C[(size_t)row * ldc + colOff + col] = v;
  }
}

// ---------------------------------------------------------------------------
// Fused attention: one node per 256-thread block.  Single pass over the
// 512 MB neighbor tensor (the HBM roofline term): mask, energies, softmax,
// and attention-weighted feature sum xbar, all from LDS.
// ---------------------------------------------------------------------------
__global__ void attn_kernel(const float* __restrict__ Xall,  // [N,K,DIN]
                            const float* __restrict__ QK,    // [N,DIN]  (Q @ Wk)
                            const float* __restrict__ Q,     // [N,DOUT]
                            const float* __restrict__ bk,    // [DOUT]
                            float* __restrict__ xbar) {      // [N,DIN]
  const int n = blockIdx.x;
  const int t = threadIdx.x;   // 0..255
  __shared__ float sX[KNBR * DIN];     // 32 KB
  __shared__ float sQK[DIN];
  __shared__ float redA[256];
  __shared__ float redB[256];
  __shared__ float sE[KNBR];
  __shared__ float sAtt[KNBR];

  const float* X = Xall + (size_t)n * KNBR * DIN;
  for (int i = t; i < (KNBR * DIN) / 4; i += 256)
    ((float4*)sX)[i] = ((const float4*)X)[i];
  sQK[t] = QK[(size_t)n * DIN + t];
  redA[t] = bk[t] * Q[(size_t)n * DOUT + t];     // c = bk . q
  __syncthreads();
  for (int s = 128; s > 0; s >>= 1) {
    if (t < s) redA[t] += redA[t + s];
    __syncthreads();
  }
  const float cval = redA[0];
  __syncthreads();

  // energies + padding mask: 8 threads per neighbor row
  const int row = t >> 3, g = t & 7;
  {
    float e = 0.f, rs = 0.f;
    const float* xr = sX + row * DIN + g * 32;
    const float* qk = sQK + g * 32;
#pragma unroll 8
    for (int j = 0; j < 32; ++j) { float x = xr[j]; e += x * qk[j]; rs += x; }
    redA[t] = e; redB[t] = rs;
  }
  __syncthreads();
  for (int s = 4; s > 0; s >>= 1) {
    if (g < s) { redA[t] += redA[t + s]; redB[t] += redB[t + s]; }
    __syncthreads();
  }
  if (g == 0) {
    // faithful to reference: masked (all-zero) rows get energy 1e-12, not -inf
    sE[row] = (redB[t] == 0.f) ? 1e-12f : (redA[t] + cval);
  }
  __syncthreads();

  if (t < KNBR) {                      // redundant per-lane softmax (cheap)
    float m = -3.4e38f;
    for (int k = 0; k < KNBR; ++k) m = fmaxf(m, sE[k]);
    float den = 0.f;
    for (int k = 0; k < KNBR; ++k) den += __expf(sE[k] - m);
    sAtt[t] = __expf(sE[t] - m) / den;
  }
  __syncthreads();

  float accv = 0.f;                     // xbar[d=t] = Σ_k att_k * X[k,d]
#pragma unroll 8
  for (int k = 0; k < KNBR; ++k) accv += sAtt[k] * sX[k * DIN + t];
  xbar[(size_t)n * DIN + t] = accv;
}

// ---------------------------------------------------------------------------
// Deterministic two-stage BatchNorm statistics over output [N, 512]
// ---------------------------------------------------------------------------
__global__ void bn_partial(const float* __restrict__ X,
                           float* __restrict__ PS, float* __restrict__ PS2) {
  const int cb = blockIdx.x;           // 0..15 : 32-channel block
  const int rb = blockIdx.y;           // 0..63 : 256-row block
  const int t  = threadIdx.x;
  const int c  = cb * 32 + (t & 31);
  const int r0 = rb * 256 + (t >> 5);
  float s = 0.f, s2 = 0.f;
#pragma unroll 4
  for (int i = 0; i < 32; ++i) {
    float v = X[(size_t)(r0 + i * 8) * DCAT + c];
    s += v; s2 += v * v;
  }
  __shared__ float A[256], B[256];
  A[t] = s; B[t] = s2; __syncthreads();
  for (int st = 128; st >= 32; st >>= 1) {
    if (t < st) { A[t] += A[t + st]; B[t] += B[t + st]; }
    __syncthreads();
  }
  if (t < 32) { PS[(size_t)rb * DCAT + c] = A[t]; PS2[(size_t)rb * DCAT + c] = B[t]; }
}

__global__ void bn_final(const float* __restrict__ PS, const float* __restrict__ PS2,
                         float* __restrict__ mean, float* __restrict__ istd) {
  const int c = blockIdx.x * 256 + threadIdx.x;   // 512 channels total
  float s = 0.f, s2 = 0.f;
  for (int i = 0; i < 64; ++i) { s += PS[i * DCAT + c]; s2 += PS2[i * DCAT + c]; }
  const float m   = s * (1.f / N_NODES);
  const float var = s2 * (1.f / N_NODES) - m * m;   // biased, as in reference
  mean[c] = m;
  istd[c] = rsqrtf(var + BN_EPS);
}

__global__ void finalize_kernel(const float* __restrict__ outc,
                                const float* __restrict__ gate,
                                const float* __restrict__ mean,
                                const float* __restrict__ istd,
                                const float* __restrict__ gamma,
                                const float* __restrict__ beta,
                                float* __restrict__ res) {
  const size_t i = (size_t)blockIdx.x * 256 + threadIdx.x;
  const int c = (int)(i & (DCAT - 1));
  float o  = outc[i];
  float bn = gamma[c] * (o - mean[c]) * istd[c] + beta[c];
  res[i] = gate[i] * fmaxf(bn, 0.f);
}

// ---------------------------------------------------------------------------
extern "C" void kernel_launch(void* const* d_in, const int* in_sizes, int n_in,
                              void* d_out, int out_size, void* d_ws, size_t ws_size,
                              hipStream_t stream) {
  const float* input = (const float*)d_in[0];
  const float* nbr   = (const float*)d_in[1];
  const float* Wq    = (const float*)d_in[2];
  const float* bq    = (const float*)d_in[3];
  const float* Wk    = (const float*)d_in[4];
  const float* bk    = (const float*)d_in[5];
  const float* Wv    = (const float*)d_in[6];
  const float* bv    = (const float*)d_in[7];
  const float* Wno   = (const float*)d_in[8];
  const float* bno   = (const float*)d_in[9];
  const float* Wio   = (const float*)d_in[10];
  const float* bio   = (const float*)d_in[11];
  const float* Wg    = (const float*)d_in[12];
  const float* bg    = (const float*)d_in[13];
  const float* gamma = (const float*)d_in[14];
  const float* beta  = (const float*)d_in[15];
  float* out = (float*)d_out;

  float* ws   = (float*)d_ws;
  float* Q    = ws;                                  // [N,256]
  float* QK   = Q    + (size_t)N_NODES * DOUT;       // [N,256]
  float* XBAR = QK   + (size_t)N_NODES * DIN;        // [N,256]
  float* CTX  = XBAR + (size_t)N_NODES * DIN;        // [N,256]
  float* OUT2 = CTX  + (size_t)N_NODES * DOUT;       // [N,512] concat
  float* GATE = OUT2 + (size_t)N_NODES * DCAT;       // [N,512]
  float* PS   = GATE + (size_t)N_NODES * DCAT;       // [64,512]
  float* PS2  = PS   + (size_t)64 * DCAT;            // [64,512]
  float* MEAN = PS2  + (size_t)64 * DCAT;            // [512]
  float* ISTD = MEAN + DCAT;                         // [512]

  const dim3 blk(256);
  const dim3 g256x8(N_NODES / 64, DOUT / 32);
  const dim3 g256x16(N_NODES / 64, DCAT / 32);

  // 1) Q = input @ Wq^T + bq
  gemm_bf16_wmma<true, false><<<g256x8, blk, 0, stream>>>(input, DIN, Wq, bq, Q, DOUT, 0, DOUT, DIN);
  // 2) QK = Q @ Wk   (bk folded into scalar c inside attention)
  gemm_bf16_wmma<false, false><<<g256x8, blk, 0, stream>>>(Q, DOUT, Wk, nullptr, QK, DIN, 0, DIN, DOUT);
  // 3) fused mask/energies/softmax/weighted-sum over neighbors (512 MB pass)
  attn_kernel<<<dim3(N_NODES), blk, 0, stream>>>(nbr, QK, Q, bk, XBAR);
  // 4) context = xbar @ Wv^T + bv
  gemm_bf16_wmma<true, false><<<g256x8, blk, 0, stream>>>(XBAR, DIN, Wv, bv, CTX, DOUT, 0, DOUT, DIN);
  // 5) output[:, :256]  = input @ Wio^T + bio
  gemm_bf16_wmma<true, false><<<g256x8, blk, 0, stream>>>(input, DIN, Wio, bio, OUT2, DCAT, 0, DOUT, DIN);
  // 6) output[:, 256:]  = context @ Wno^T + bno
  gemm_bf16_wmma<true, false><<<g256x8, blk, 0, stream>>>(CTX, DOUT, Wno, bno, OUT2, DCAT, DOUT, DOUT, DOUT);
  // 7) gate = relu(output @ Wg^T + bg)
  gemm_bf16_wmma<true, true><<<g256x16, blk, 0, stream>>>(OUT2, DCAT, Wg, bg, GATE, DCAT, 0, DCAT, DCAT);
  // 8) BatchNorm batch statistics (deterministic two-stage)
  bn_partial<<<dim3(16, 64), blk, 0, stream>>>(OUT2, PS, PS2);
  bn_final<<<dim3(2), blk, 0, stream>>>(PS, PS2, MEAN, ISTD);
  // 9) result = gate * relu(bn(output))
  finalize_kernel<<<dim3((N_NODES * DCAT) / 256), blk, 0, stream>>>(OUT2, GATE, MEAN, ISTD, gamma, beta, out);
}
